// w2v_model_11836929868572
// MI455X (gfx1250) — matrix operand order repaired and checked by reading, not compiled, not run on hardware
//
#include <hip/hip_runtime.h>
#include <hip/hip_bf16.h>

typedef unsigned int u32;
typedef unsigned short u16;
typedef __attribute__((ext_vector_type(16))) __bf16 v16bf;
typedef __attribute__((ext_vector_type(8)))  float  v8f;
typedef __attribute__((ext_vector_type(4)))  u32    v4u;

#define B_SZ   4096
#define S_SZ   10
#define H_SZ   32
#define DH_SZ  64
#define E_SZ   1024
#define HID_SZ 2048
#define N_TOT  6144

union FragBF { v4u q[2]; v16bf f; };

__device__ __forceinline__ u16 f2bf(float x) {
  union { float f; u32 u; } v; v.f = x;
  u32 r = v.u + 0x7FFFu + ((v.u >> 16) & 1u);   // round-to-nearest-even
  return (u16)(r >> 16);
}

// ---------- pack kernels ----------
__global__ void gather_bf16_kernel(const int* __restrict__ t,
                                   const float* __restrict__ emb,
                                   u16* __restrict__ Abf) {
  int i = blockIdx.x * blockDim.x + threadIdx.x;
  if (i >= B_SZ * E_SZ) return;
  int b = i >> 10, e = i & 1023;
  Abf[i] = f2bf(emb[(long long)t[b] * E_SZ + e]);
}

__global__ void pack_w_kernel(const float* __restrict__ Wq,
                              const float* __restrict__ Wk,
                              const float* __restrict__ Wv,
                              u16* __restrict__ Wbf) {
  int i = blockIdx.x * blockDim.x + threadIdx.x;
  if (i >= HID_SZ * E_SZ) return;
  int which = blockIdx.y;
  const float* src = (which == 0) ? Wq : (which == 1) ? Wk : Wv;
  Wbf[(size_t)which * (HID_SZ * E_SZ) + i] = f2bf(src[i]);
}

__global__ void pack_bias_kernel(const float* __restrict__ bq,
                                 const float* __restrict__ bk,
                                 const float* __restrict__ bv,
                                 float* __restrict__ biasc) {
  int i = blockIdx.x * blockDim.x + threadIdx.x;
  if (i >= N_TOT) return;
  biasc[i] = (i < HID_SZ) ? bq[i] : (i < 2 * HID_SZ) ? bk[i - HID_SZ] : bv[i - 2 * HID_SZ];
}

// ---------- fused QKV GEMM: C[4096,6144] = Abf x Wbf^T + bias ----------
// block = 256 thr (8 waves); wave -> 16(M) x 64(N) tile; K-step 32 (bf16 WMMA)
__global__ __launch_bounds__(256) void qkv_gemm_kernel(
    const u16* __restrict__ Abf, const u16* __restrict__ Wbf,
    const float* __restrict__ biasc,
    u16* __restrict__ Qbf, u16* __restrict__ Kbf, float* __restrict__ Vout) {
  const int lane = threadIdx.x & 31;
  const int wave = threadIdx.x >> 5;
  const int m0 = blockIdx.y * 128 + wave * 16;
  const int n0 = blockIdx.x * 64;
  const int l15 = lane & 15;
  const int hi  = lane >> 4;          // lane-half select

  v8f acc[4] = {};
  const u16* arow = Abf + (size_t)(m0 + l15) * E_SZ;
  const int aoff = hi * 8;            // A: K {0-7,16-23} vs {8-15,24-31}

  for (int kb = 0; kb < E_SZ; kb += 32) {
    FragBF a;
    a.q[0] = *(const v4u*)(arow + kb + aoff);
    a.q[1] = *(const v4u*)(arow + kb + aoff + 16);
#pragma unroll
    for (int t = 0; t < 4; ++t) {
      const u16* brow = Wbf + (size_t)(n0 + t * 16 + l15) * E_SZ + kb + hi * 8;
      FragBF bm;
      bm.q[0] = *(const v4u*)(brow);
      bm.q[1] = *(const v4u*)(brow + 16);
      acc[t] = __builtin_amdgcn_wmma_f32_16x16x32_bf16(
          false, a.f, false, bm.f, (short)0, acc[t], false, false);
    }
  }

#pragma unroll
  for (int t = 0; t < 4; ++t) {
    int col = n0 + t * 16 + l15;
    float bias = biasc[col];
    if (col < HID_SZ) {
#pragma unroll
      for (int r = 0; r < 8; ++r) {
        int row = m0 + r + hi * 8;
        Qbf[(size_t)row * HID_SZ + col] = f2bf(acc[t][r] + bias);
      }
    } else if (col < 2 * HID_SZ) {
      int c2 = col - HID_SZ;
#pragma unroll
      for (int r = 0; r < 8; ++r) {
        int row = m0 + r + hi * 8;
        Kbf[(size_t)row * HID_SZ + c2] = f2bf(acc[t][r] + bias);
      }
    } else {
      int c2 = col - 2 * HID_SZ;
#pragma unroll
      for (int r = 0; r < 8; ++r) {
        int row = m0 + r + hi * 8;
        Vout[(size_t)row * HID_SZ + c2] = acc[t][r] + bias;   // target_vector
      }
    }
  }
}

// ---------- Vsum^T: Vsum[g][d] = sum_{s<10} target_vector[s][g*64+d]; store [d][g] bf16 ----------
__global__ void vsum_kernel(const float* __restrict__ Vout, u16* __restrict__ VsumT) {
  int c = blockIdx.x * blockDim.x + threadIdx.x;
  if (c >= HID_SZ) return;
  float s = 0.f;
#pragma unroll
  for (int m = 0; m < S_SZ; ++m) s += Vout[m * HID_SZ + c];
  int h = c >> 6, d = c & 63;
  VsumT[d * H_SZ + h] = f2bf(s);
}

// ---------- per-batch attention: one wave per block ----------
__global__ __launch_bounds__(32) void attn_kernel(
    const u16* __restrict__ Qbf, const u16* __restrict__ Kbf,
    const u16* __restrict__ VsumT, float* __restrict__ ctx_out) {
  __shared__ __align__(16) u16 attn_lds[H_SZ * H_SZ];
  const int b = blockIdx.x;
  const int lane = threadIdx.x & 31;
  const int l15 = lane & 15;
  const int hi  = lane >> 4;

  const u16* qb = Qbf + (size_t)b * HID_SZ;
  const u16* kb = Kbf + (size_t)b * HID_SZ;

  // scores[32x32] = Q[32x64] @ K^T[64x32]
  v8f sc[2][2] = {};
#pragma unroll
  for (int ks = 0; ks < 2; ++ks) {
    FragBF aq[2];
#pragma unroll
    for (int mi = 0; mi < 2; ++mi) {
      const u16* p = qb + (mi * 16 + l15) * DH_SZ + ks * 32 + hi * 8;
      aq[mi].q[0] = *(const v4u*)(p);
      aq[mi].q[1] = *(const v4u*)(p + 16);
    }
#pragma unroll
    for (int ni = 0; ni < 2; ++ni) {
      const u16* p = kb + (ni * 16 + l15) * DH_SZ + ks * 32 + hi * 8;
      FragBF bk_;
      bk_.q[0] = *(const v4u*)(p);
      bk_.q[1] = *(const v4u*)(p + 16);
#pragma unroll
      for (int mi = 0; mi < 2; ++mi)
        sc[mi][ni] = __builtin_amdgcn_wmma_f32_16x16x32_bf16(
            false, aq[mi].f, false, bk_.f, (short)0, sc[mi][ni], false, false);
    }
  }

  // softmax over g (32 values per row: 2 ni-tiles x 16 lanes of a half)
  const float scale = 0.125f;  // 1/sqrt(64)
#pragma unroll
  for (int mi = 0; mi < 2; ++mi) {
#pragma unroll
    for (int r = 0; r < 8; ++r) {
      float x0 = sc[mi][0][r] * scale;
      float x1 = sc[mi][1][r] * scale;
      float mx = fmaxf(x0, x1);
#pragma unroll
      for (int msk = 1; msk <= 8; msk <<= 1) mx = fmaxf(mx, __shfl_xor(mx, msk, 32));
      float e0 = __expf(x0 - mx), e1 = __expf(x1 - mx);
      float s = e0 + e1;
#pragma unroll
      for (int msk = 1; msk <= 8; msk <<= 1) s += __shfl_xor(s, msk, 32);
      float inv = 1.0f / s;
      int row = mi * 16 + r + hi * 8;
      attn_lds[row * H_SZ + l15]      = f2bf(e0 * inv);
      attn_lds[row * H_SZ + l15 + 16] = f2bf(e1 * inv);
    }
  }
  asm volatile("s_wait_dscnt 0" ::: "memory");

  // context[32x64] = attn[32x32] @ Vsum[32x64]   (B-frags contiguous via VsumT)
  FragBF aa[2];
#pragma unroll
  for (int mi = 0; mi < 2; ++mi) {
    const u16* p = attn_lds + (mi * 16 + l15) * H_SZ + hi * 8;
    aa[mi].q[0] = *(const v4u*)(p);
    aa[mi].q[1] = *(const v4u*)(p + 16);
  }
  float* outp = ctx_out + (size_t)b * HID_SZ;
#pragma unroll
  for (int ni = 0; ni < 4; ++ni) {
    const u16* p = VsumT + (ni * 16 + l15) * H_SZ + hi * 8;
    FragBF bv_;
    bv_.q[0] = *(const v4u*)(p);
    bv_.q[1] = *(const v4u*)(p + 16);
#pragma unroll
    for (int mi = 0; mi < 2; ++mi) {
      v8f c = {};
      c = __builtin_amdgcn_wmma_f32_16x16x32_bf16(
          false, aa[mi].f, false, bv_.f, (short)0, c, false, false);
#pragma unroll
      for (int r = 0; r < 8; ++r) {
        int row = mi * 16 + r + hi * 8;
        outp[row * DH_SZ + ni * 16 + l15] = c[r];
      }
    }
  }
}

extern "C" void kernel_launch(void* const* d_in, const int* in_sizes, int n_in,
                              void* d_out, int out_size, void* d_ws, size_t ws_size,
                              hipStream_t stream) {
  (void)in_sizes; (void)n_in; (void)out_size; (void)ws_size;
  const int*   t   = (const int*)d_in[0];
  /* d_in[1] = c : unused by reference */
  const float* emb = (const float*)d_in[2];
  const float* Wq  = (const float*)d_in[3];
  const float* bq  = (const float*)d_in[4];
  const float* Wk  = (const float*)d_in[5];
  const float* bk  = (const float*)d_in[6];
  const float* Wv  = (const float*)d_in[7];
  const float* bv  = (const float*)d_in[8];

  float* out_tv = (float*)d_out;                        // target_vector [B,HID]
  float* out_cv = out_tv + (size_t)B_SZ * HID_SZ;       // context_vector [B,HID]

  char* ws = (char*)d_ws;
  u16* Abf = (u16*)ws;        ws += (size_t)B_SZ * E_SZ * 2;      // 8 MB
  u16* Wbf = (u16*)ws;        ws += (size_t)N_TOT * E_SZ * 2;     // 12 MB
  float* biasc = (float*)ws;  ws += (size_t)N_TOT * 4;            // 24 KB
  u16* Qbf = (u16*)ws;        ws += (size_t)B_SZ * HID_SZ * 2;    // 16 MB
  u16* Kbf = (u16*)ws;        ws += (size_t)B_SZ * HID_SZ * 2;    // 16 MB
  u16* VsumT = (u16*)ws;      ws += (size_t)HID_SZ * 2;           // 4 KB

  gather_bf16_kernel<<<(B_SZ * E_SZ + 255) / 256, 256, 0, stream>>>(t, emb, Abf);
  dim3 gw((HID_SZ * E_SZ + 255) / 256, 3);
  pack_w_kernel<<<gw, 256, 0, stream>>>(Wq, Wk, Wv, Wbf);
  pack_bias_kernel<<<(N_TOT + 255) / 256, 256, 0, stream>>>(bq, bk, bv, biasc);

  dim3 gg(N_TOT / 64, B_SZ / 128);
  qkv_gemm_kernel<<<gg, 256, 0, stream>>>(Abf, Wbf, biasc, Qbf, Kbf, out_tv);

  vsum_kernel<<<(HID_SZ + 255) / 256, 256, 0, stream>>>(out_tv, VsumT);
  attn_kernel<<<B_SZ, 32, 0, stream>>>(Qbf, Kbf, VsumT, out_cv);
}
